// GNN_69028714381395
// MI455X (gfx1250) — matrix operand order repaired and checked by reading, not compiled, not run on hardware
//
#include <hip/hip_runtime.h>
#include <hip/hip_bf16.h>

typedef _Float16 h16;
typedef __attribute__((ext_vector_type(16))) _Float16 v16h;
typedef __attribute__((ext_vector_type(8)))  _Float16 v8h;
typedef __attribute__((ext_vector_type(4)))  _Float16 v4h;
typedef __attribute__((ext_vector_type(8)))  float    v8f;

#define D64 64
#define NLAYERS 5
#define LSTR 193          // LDS row stride (odd -> breaks bank conflicts)
#define SCAN_CHUNK 2048   // elements per scan1 block (256 thr x 8)

// ---------------------------------------------------------------------------
// WMMA tile loaders (v_wmma_f32_16x16x32_f16; A/B use this code's contiguous k order)
// A 16x32 f16: lane&15 = M; element j -> K = (lane>>4)*16 + j
//   -> two contiguous 8-half (16B) runs per lane.
// B 32x16 f16: lane&15 = N; element j -> K = (lane>>4)*16 + j (contiguous 32B)
//   -> weights stored transposed wt[col*64 + k].
// C/D 16x16 f32: lane&15 = N; element r -> M = r + 8*(lane>>4).
// ---------------------------------------------------------------------------
__device__ __forceinline__ v16h load_a_tile(const h16* __restrict__ base, int row,
                                            int kb, int kHalf) {
  const h16* p = base + row * D64 + kb + kHalf * 16;
  union { v16h v; v8h h[2]; } u;
  u.h[0] = *(const v8h*)(p);
  u.h[1] = *(const v8h*)(p + 8);
  return u.v;
}

__device__ __forceinline__ v16h load_b_tile(const h16* __restrict__ wt, int col,
                                            int kb, int kHalf) {
  return *(const v16h*)(wt + col * D64 + kb + kHalf * 16);
}

// ---------------------------------------------------------------------------
// Weight prep: f32 -> f16; conv weight transposed to [d][k]; GRU weights
// (3D, D) row-major are already the transposed-B layout.
// ---------------------------------------------------------------------------
__global__ void prep_weights_kernel(const float* __restrict__ cw,
                                    const float* __restrict__ wih,
                                    const float* __restrict__ whh,
                                    h16* __restrict__ cwt16,
                                    h16* __restrict__ wih16,
                                    h16* __restrict__ whh16) {
  int i = blockIdx.x * blockDim.x + threadIdx.x;
  const int CW = NLAYERS * D64 * D64;
  const int GW = NLAYERS * 3 * D64 * D64;
  if (i < CW) {
    int l = i / (D64 * D64), r = i % (D64 * D64);
    int k = r / D64, d = r % D64;
    cwt16[l * D64 * D64 + d * D64 + k] = (h16)cw[i];
  }
  if (i < GW) {
    wih16[i] = (h16)wih[i];
    whh16[i] = (h16)whh[i];
  }
}

__global__ void zero16_kernel(float4* __restrict__ p, int n) {
  int i = blockIdx.x * blockDim.x + threadIdx.x;
  if (i < n) p[i] = make_float4(0.f, 0.f, 0.f, 0.f);
}

// ---------------------------------------------------------------------------
// CSR build: counting sort of edges by dst (built once, reused 4x).
// ---------------------------------------------------------------------------
__global__ void count_kernel(const int* __restrict__ ei, int* __restrict__ deg, int E) {
  int e = blockIdx.x * blockDim.x + threadIdx.x;
  if (e < E) atomicAdd(&deg[ei[E + e]], 1);
}

__global__ void scan1_kernel(const int* __restrict__ deg, int* __restrict__ out,
                             int* __restrict__ blockSums, int N) {
  __shared__ int sh[256];
  int t = threadIdx.x;
  int idx0 = blockIdx.x * SCAN_CHUNK + t * 8;
  int vals[8];
  int sum = 0;
#pragma unroll
  for (int i = 0; i < 8; ++i) {
    int v = (idx0 + i < N) ? deg[idx0 + i] : 0;
    vals[i] = sum;
    sum += v;
  }
  sh[t] = sum;
  __syncthreads();
  for (int off = 1; off < 256; off <<= 1) {
    int v = (t >= off) ? sh[t - off] : 0;
    __syncthreads();
    sh[t] += v;
    __syncthreads();
  }
  int thrExcl = (t == 0) ? 0 : sh[t - 1];
#pragma unroll
  for (int i = 0; i < 8; ++i)
    if (idx0 + i < N) out[idx0 + i] = thrExcl + vals[i];
  if (t == 255) blockSums[blockIdx.x] = sh[255];
}

__global__ void scan2_kernel(int* __restrict__ blockSums, int nb) {
  if (threadIdx.x == 0) {
    int run = 0;
    for (int i = 0; i < nb; ++i) { int v = blockSums[i]; blockSums[i] = run; run += v; }
  }
}

__global__ void scan3_kernel(int* __restrict__ offs, const int* __restrict__ blockSums,
                             int* __restrict__ cursor, int N, int E) {
  int i = blockIdx.x * blockDim.x + threadIdx.x;
  if (i < N) {
    int v = offs[i] + blockSums[i / SCAN_CHUNK];
    offs[i] = v;
    cursor[i] = v;
  } else if (i == N) {
    offs[N] = E;
  }
}

__global__ void fill_kernel(const int* __restrict__ ei, int* __restrict__ cursor,
                            int* __restrict__ esrc, int E) {
  int e = blockIdx.x * blockDim.x + threadIdx.x;
  if (e < E) {
    int d = ei[E + e];
    int pos = atomicAdd(&cursor[d], 1);
    esrc[pos] = ei[e];
  }
}

// ---------------------------------------------------------------------------
// m16 = x16 @ W. 4 waves/block, one 16-col tile per wave, 16 rows/block.
// ---------------------------------------------------------------------------
__global__ void mul_w_kernel(const h16* __restrict__ x16, const h16* __restrict__ wt,
                             h16* __restrict__ m16, int N) {
  int lane = threadIdx.x & 31;
  int wave = threadIdx.x >> 5;
  int rowBase = blockIdx.x * 16;
  int colB = lane & 15, kHalf = lane >> 4;
  int col = wave * 16 + colB;
  int arow = rowBase + colB; if (arow >= N) arow = N - 1;

  v8f acc = {};
#pragma unroll
  for (int kb = 0; kb < D64; kb += 32) {
    v16h a = load_a_tile(x16, arow, kb, kHalf);
    v16h b = load_b_tile(wt, col, kb, kHalf);
    acc = __builtin_amdgcn_wmma_f32_16x16x32_f16(false, a, false, b, (short)0, acc,
                                                 false, false);
  }
  if (rowBase + 16 <= N) {         // uniform fast path: no per-row predication
#pragma unroll
    for (int r = 0; r < 8; ++r)
      m16[(rowBase + r + kHalf * 8) * D64 + col] = (h16)acc[r];
  } else {
#pragma unroll
    for (int r = 0; r < 8; ++r) {
      int row = rowBase + r + kHalf * 8;
      if (row < N) m16[row * D64 + col] = (h16)acc[r];
    }
  }
}

// ---------------------------------------------------------------------------
// Atomic-free gather: agg16[n] = sum_{e in CSR[n]} m16[esrc[e]].
// One wave per node. Per 32-edge chunk: one coalesced 128B load of the edge
// indices, then ds_bpermute broadcasts; two 16-lane halves take alternating
// edges with 8B (v4h) row loads, unrolled x2 so 4 row loads are in flight.
// Cross-half reduce via shfl_xor(16); one 128B row store per node.
// m16 (12.8MB) is L2-resident, so this runs at L2 bandwidth.
// ---------------------------------------------------------------------------
__global__ void gather_kernel(const int* __restrict__ offs, const int* __restrict__ esrc,
                              const h16* __restrict__ m16, h16* __restrict__ agg16,
                              int N) {
  int node = (blockIdx.x * blockDim.x + threadIdx.x) >> 5;
  int lane = threadIdx.x & 31;
  if (node >= N) return;
  int s0 = offs[node], s1 = offs[node + 1];
  int half = lane >> 4, li = lane & 15;
  const h16* mrow = m16 + li * 4;
  float a0 = 0.f, a1 = 0.f, a2 = 0.f, a3 = 0.f;

  for (int base = s0; base < s1; base += 32) {
    int cnt = s1 - base; if (cnt > 32) cnt = 32;
    int idx = (lane < cnt) ? esrc[base + lane] : 0;   // coalesced index load
    int i = half;
    for (; i + 2 < cnt; i += 4) {                     // 2 edges per half per iter
      int src0 = __shfl(idx, i, 32);
      int src1 = __shfl(idx, i + 2, 32);
      v4h m0 = *(const v4h*)(mrow + src0 * D64);
      v4h m1 = *(const v4h*)(mrow + src1 * D64);
      a0 += (float)m0.x + (float)m1.x;
      a1 += (float)m0.y + (float)m1.y;
      a2 += (float)m0.z + (float)m1.z;
      a3 += (float)m0.w + (float)m1.w;
    }
    for (; i < cnt; i += 2) {
      int src = __shfl(idx, i, 32);
      v4h mv = *(const v4h*)(mrow + src * D64);
      a0 += (float)mv.x; a1 += (float)mv.y;
      a2 += (float)mv.z; a3 += (float)mv.w;
    }
  }
  a0 += __shfl_xor(a0, 16, 32);
  a1 += __shfl_xor(a1, 16, 32);
  a2 += __shfl_xor(a2, 16, 32);
  a3 += __shfl_xor(a3, 16, 32);
  if (half == 0) {
    v4h o = { (h16)a0, (h16)a1, (h16)a2, (h16)a3 };
    *(v4h*)(agg16 + node * D64 + li * 4) = o;
  }
}

// ---------------------------------------------------------------------------
// GRU cell: gi = agg16 @ w_ih^T, gh = x16 @ w_hh^T via WMMA into LDS, then
// f32 gate math; updates x (f32) and x16 (f16) in place.
// ---------------------------------------------------------------------------
__global__ void gru_kernel(const h16* __restrict__ agg16,
                           const h16* __restrict__ wih16, const h16* __restrict__ whh16,
                           const float* __restrict__ bih, const float* __restrict__ bhh,
                           float* __restrict__ x, h16* __restrict__ x16, int N) {
  __shared__ float gi[16 * LSTR];
  __shared__ float gh[16 * LSTR];
  int lane = threadIdx.x & 31;
  int wave = threadIdx.x >> 5;              // 0..7
  int rowBase = blockIdx.x * 16;
  int colB = lane & 15, kHalf = lane >> 4;
  int arow = rowBase + colB; if (arow >= N) arow = N - 1;
  bool full = (rowBase + 16 <= N);

  for (int t = wave; t < 24; t += 8) {
    bool isGh = t >= 12;
    int ct = isGh ? (t - 12) : t;
    const h16* src = isGh ? x16 : agg16;
    const h16* wt  = isGh ? whh16 : wih16;
    float* out     = isGh ? gh : gi;
    v8f acc = {};
#pragma unroll
    for (int kb = 0; kb < D64; kb += 32) {
      v16h a = load_a_tile(src, arow, kb, kHalf);
      v16h b = load_b_tile(wt, ct * 16 + colB, kb, kHalf);
      acc = __builtin_amdgcn_wmma_f32_16x16x32_f16(false, a, false, b, (short)0, acc,
                                                   false, false);
    }
#pragma unroll
    for (int r = 0; r < 8; ++r)
      out[(r + kHalf * 8) * LSTR + ct * 16 + colB] = acc[r];
  }
  __syncthreads();

#pragma unroll
  for (int i = 0; i < 4; ++i) {
    int idx = threadIdx.x + i * 256;
    int mrow = idx >> 6, col = idx & 63;
    int row = rowBase + mrow;
    if (full || row < N) {
      float giR = gi[mrow * LSTR + col]       + bih[col];
      float giZ = gi[mrow * LSTR + 64 + col]  + bih[64 + col];
      float giN = gi[mrow * LSTR + 128 + col] + bih[128 + col];
      float ghR = gh[mrow * LSTR + col]       + bhh[col];
      float ghZ = gh[mrow * LSTR + 64 + col]  + bhh[64 + col];
      float ghN = gh[mrow * LSTR + 128 + col] + bhh[128 + col];
      float rg = 1.f / (1.f + __expf(-(giR + ghR)));
      float zg = 1.f / (1.f + __expf(-(giZ + ghZ)));
      float ng = tanhf(giN + rg * ghN);
      float h  = x[row * D64 + col];
      float hn = (1.f - zg) * ng + zg * h;
      x[row * D64 + col]   = hn;
      x16[row * D64 + col] = (h16)hn;
    }
  }
}

// ---------------------------------------------------------------------------
// Pool: LDS-staged segment-sum over (sorted) batch -> g[64][64].
// ---------------------------------------------------------------------------
#define POOL_CHUNK 2048
__global__ void pool_kernel(const float* __restrict__ x, const int* __restrict__ batch,
                            float* __restrict__ g, int N, int G) {
  __shared__ float acc[64 * 64];
  for (int i = threadIdx.x; i < 64 * 64; i += blockDim.x) acc[i] = 0.f;
  __syncthreads();
  int base = blockIdx.x * POOL_CHUNK;
  int nmax = N - base; if (nmax > POOL_CHUNK) nmax = POOL_CHUNK;
  for (int idx = threadIdx.x; idx < nmax * 64; idx += blockDim.x) {
    int n = base + (idx >> 6), c = idx & 63;
    atomicAdd(&acc[batch[n] * 64 + c], x[n * D64 + c]);
  }
  __syncthreads();
  int gmax = G * 64;
  for (int i = threadIdx.x; i < gmax; i += blockDim.x)
    __hip_atomic_fetch_add(&g[i], acc[i], __ATOMIC_RELAXED, __HIP_MEMORY_SCOPE_AGENT);
}

// ---------------------------------------------------------------------------
// Final MLP: G graphs x (64->32->16->1) with ELU.
// ---------------------------------------------------------------------------
__global__ void mlp_kernel(const float* __restrict__ g,
                           const float* __restrict__ fc1w, const float* __restrict__ fc1b,
                           const float* __restrict__ fc2w, const float* __restrict__ fc2b,
                           const float* __restrict__ fc3w, const float* __restrict__ fc3b,
                           float* __restrict__ out, int G) {
  int gid = blockIdx.x * blockDim.x + threadIdx.x;
  if (gid >= G) return;
  const float* gr = g + gid * 64;
  float h1[32], h2[16];
#pragma unroll
  for (int j = 0; j < 32; ++j) {
    float s = fc1b[j];
    for (int k = 0; k < 64; ++k) s += gr[k] * fc1w[j * 64 + k];
    h1[j] = (s > 0.f) ? s : (__expf(s) - 1.f);
  }
#pragma unroll
  for (int j = 0; j < 16; ++j) {
    float s = fc2b[j];
    for (int k = 0; k < 32; ++k) s += h1[k] * fc2w[j * 32 + k];
    h2[j] = (s > 0.f) ? s : (__expf(s) - 1.f);
  }
  float s = fc3b[0];
#pragma unroll
  for (int k = 0; k < 16; ++k) s += h2[k] * fc3w[k];
  out[gid] = s;
}

// ---------------------------------------------------------------------------
extern "C" void kernel_launch(void* const* d_in, const int* in_sizes, int n_in,
                              void* d_out, int out_size, void* d_ws, size_t ws_size,
                              hipStream_t stream) {
  const float* conv_w = (const float*)d_in[0];
  const float* wih    = (const float*)d_in[1];
  const float* whh    = (const float*)d_in[2];
  const float* bih    = (const float*)d_in[3];
  const float* bhh    = (const float*)d_in[4];
  const float* fc1w   = (const float*)d_in[5];
  const float* fc1b   = (const float*)d_in[6];
  const float* fc2w   = (const float*)d_in[7];
  const float* fc2b   = (const float*)d_in[8];
  const float* fc3w   = (const float*)d_in[9];
  const float* fc3b   = (const float*)d_in[10];
  const int*   eidx   = (const int*)d_in[11];
  const int*   batch  = (const int*)d_in[12];

  const int N = in_sizes[12];
  const int E = in_sizes[11] / 2;
  const int G = out_size;

  // ---- workspace carve (256B aligned) ----
  char* p = (char*)d_ws;
  auto carve = [&](size_t bytes) -> void* {
    void* r = (void*)p;
    p += (bytes + 255) & ~(size_t)255;
    return r;
  };
  float* x      = (float*)carve((size_t)N * D64 * 4);
  h16*   x16    = (h16*)  carve((size_t)N * D64 * 2);
  h16*   agg16  = (h16*)  carve((size_t)N * D64 * 2);
  h16*   m16    = (h16*)  carve((size_t)N * D64 * 2);
  h16*   cwt16  = (h16*)  carve((size_t)NLAYERS * D64 * D64 * 2);
  h16*   wih16  = (h16*)  carve((size_t)NLAYERS * 3 * D64 * D64 * 2);
  h16*   whh16  = (h16*)  carve((size_t)NLAYERS * 3 * D64 * D64 * 2);
  int*   deg    = (int*)  carve((size_t)N * 4);
  int*   offs   = (int*)  carve((size_t)(N + 1) * 4);
  int*   cursor = (int*)  carve((size_t)N * 4);
  int*   esrc   = (int*)  carve((size_t)E * 4);
  int*   bsums  = (int*)  carve((size_t)4096);
  float* gpool  = (float*)carve((size_t)64 * 64 * 4);

  const int rowBlocks = (N + 15) / 16;
  const int NB = (N + SCAN_CHUNK - 1) / SCAN_CHUNK;

  // 1) weights -> f16
  {
    int n = NLAYERS * 3 * D64 * D64;
    prep_weights_kernel<<<(n + 255) / 256, 256, 0, stream>>>(conv_w, wih, whh,
                                                             cwt16, wih16, whh16);
  }
  // 2) zero state (layer-0 inputs are all zero) + zero degree counters
  {
    int nx = N * 16, nh = N * 8, nd = (N + 3) / 4;
    zero16_kernel<<<(nx + 255) / 256, 256, 0, stream>>>((float4*)x, nx);
    zero16_kernel<<<(nh + 255) / 256, 256, 0, stream>>>((float4*)x16, nh);
    zero16_kernel<<<(nh + 255) / 256, 256, 0, stream>>>((float4*)agg16, nh);
    zero16_kernel<<<(nd + 255) / 256, 256, 0, stream>>>((float4*)deg, nd);
  }
  // 3) build CSR by dst (once; reused for all layers)
  count_kernel<<<(E + 255) / 256, 256, 0, stream>>>(eidx, deg, E);
  scan1_kernel<<<NB, 256, 0, stream>>>(deg, offs, bsums, N);
  scan2_kernel<<<1, 32, 0, stream>>>(bsums, NB);
  scan3_kernel<<<(N + 1 + 255) / 256, 256, 0, stream>>>(offs, bsums, cursor, N, E);
  fill_kernel<<<(E + 255) / 256, 256, 0, stream>>>(eidx, cursor, esrc, E);
  // 4) layer 0: x==0 => agg==0; just the GRU on zero inputs
  gru_kernel<<<rowBlocks, 256, 0, stream>>>(agg16, wih16, whh16, bih, bhh, x, x16, N);
  // 5) layers 1..4
  for (int l = 1; l < NLAYERS; ++l) {
    mul_w_kernel<<<rowBlocks, 128, 0, stream>>>(x16, cwt16 + l * D64 * D64, m16, N);
    gather_kernel<<<(N * 32 + 255) / 256, 256, 0, stream>>>(offs, esrc, m16, agg16, N);
    gru_kernel<<<rowBlocks, 256, 0, stream>>>(agg16,
                                              wih16 + l * 3 * D64 * D64,
                                              whh16 + l * 3 * D64 * D64,
                                              bih + l * 3 * D64, bhh + l * 3 * D64,
                                              x, x16, N);
  }
  // 6) pool + MLP
  zero16_kernel<<<(1024 + 255) / 256, 256, 0, stream>>>((float4*)gpool, 1024);
  pool_kernel<<<(N + POOL_CHUNK - 1) / POOL_CHUNK, 256, 0, stream>>>(x, batch, gpool, N, G);
  mlp_kernel<<<(G + 63) / 64, 64, 0, stream>>>(gpool, fc1w, fc1b, fc2w, fc2b,
                                               fc3w, fc3b, (float*)d_out, G);
}
